// BayesianResNet_71408126263673
// MI455X (gfx1250) — matrix operand order repaired and checked by reading, not compiled, not run on hardware
//
#include <hip/hip_runtime.h>

// ---- CDNA5 WMMA types ------------------------------------------------------
typedef __attribute__((ext_vector_type(16))) __bf16 v16bf;
typedef __attribute__((ext_vector_type(8)))  float  v8f;
typedef __attribute__((ext_vector_type(4)))  int    i4x;   // trivial 16B vector

#define LDH 40   // LDS row stride in halves (32 + 8 pad = 80B -> 20-bank step, conflict-free)

#if __has_builtin(__builtin_amdgcn_global_load_async_to_lds_b128)
#define HAVE_ASYNC 1
// 16B global -> LDS copy through the async path (ASYNCcnt, no VGPR round-trip).
// Builtin signature (from compiler diagnostic): (v4i AS1*, v4i AS3*, imm off, imm cpol)
__device__ __forceinline__ void async_b128(const void* g, void* l) {
    __builtin_amdgcn_global_load_async_to_lds_b128(
        (__attribute__((address_space(1))) i4x*)(g),
        (__attribute__((address_space(3))) i4x*)(l),
        /*offset=*/0, /*cpol=*/0);
}
__device__ __forceinline__ void wait_async0() {
#if __has_builtin(__builtin_amdgcn_s_wait_asynccnt)
    __builtin_amdgcn_s_wait_asynccnt(0);
#else
    asm volatile("s_wait_asynccnt 0x0" ::: "memory");
#endif
}
#else
#define HAVE_ASYNC 0
#endif

__device__ __forceinline__ unsigned short f2bf(float f) {   // round-to-nearest-even
    unsigned u = __float_as_uint(f);
    u += 0x7fffu + ((u >> 16) & 1u);
    return (unsigned short)(u >> 16);
}

// ---- prepass 1: weight f32 [i][oc][c][3][3] -> bf16 [i][oc][k], k = tap*128 + c
__global__ __launch_bounds__(256)
void prep_weight(const float* __restrict__ w, unsigned short* __restrict__ wbf) {
    const int row = blockIdx.x;                       // i*128 + oc  (4096 rows)
    const float* src = w + (size_t)row * 1152;
    unsigned short* dst = wbf + (size_t)row * 1152;
    for (int k = threadIdx.x; k < 1152; k += 256) {
        const int tap = k >> 7, c = k & 127;
        dst[k] = f2bf(src[c * 9 + tap]);              // coalesced writes, stride-9 gather
    }
}

// ---- prepass 2: x f32 NCHW -> bf16 NHWC zero-padded  xT[i][pos=34*34][c=128]
// (border comes pre-zeroed via hipMemsetAsync). LDS 32x32 tile keeps both the
// f32 reads and the bf16 writes coalesced.
__global__ __launch_bounds__(256)
void prep_xT(const float* __restrict__ x, unsigned short* __restrict__ xT) {
    __shared__ unsigned short t[32][33];
    const int i  = blockIdx.y;                        // sample
    const int cb = blockIdx.x * 32;                   // channel block
    const float* src = x + ((size_t)i * 128 + cb) * 1024;
    unsigned short* dst = xT + (size_t)i * 1156 * 128 + cb;
    for (int sp0 = 0; sp0 < 1024; sp0 += 32) {
        __syncthreads();
        #pragma unroll
        for (int r = 0; r < 4; ++r) {                 // read 32c x 32px, coalesced in px
            const int e = threadIdx.x + r * 256;
            const int cl = e >> 5, j = e & 31;
            t[cl][j] = f2bf(src[(size_t)cl * 1024 + sp0 + j]);
        }
        __syncthreads();
        #pragma unroll
        for (int r = 0; r < 4; ++r) {                 // write 32px x 32c, 64B contiguous
            const int e = threadIdx.x + r * 256;
            const int j = e >> 5, cl = e & 31;
            const int sp = sp0 + j;
            const size_t pos = (size_t)((sp >> 5) + 1) * 34 + (sp & 31) + 1;
            dst[pos * 128 + cl] = t[cl][j];
        }
    }
}

// ---- main implicit-GEMM conv: per sample  C[128 oc, 1024 pix] = A[128,1152] x B[1152,1024]
// K tap-major (k = tap*128 + c). Block = 128oc x 128pix tile, 8 wave32.
// Double-buffered LDS; staging is 4 async b128 copies per thread per chunk.
template <bool PRE>
__global__ __launch_bounds__(256)
void conv3x3_wmma(const float* __restrict__ x,             // raw f32 (fallback)
                  const float* __restrict__ w,             // raw f32 (fallback)
                  const unsigned short* __restrict__ xT,   // bf16 NHWC padded (PRE)
                  const unsigned short* __restrict__ wbf,  // bf16 GEMM weights (PRE)
                  const float* __restrict__ bias,
                  float* __restrict__ out)
{
    __shared__ __align__(16) unsigned short As[2][128 * LDH]; // [oc][kloc]
    __shared__ __align__(16) unsigned short Bs[2][128 * LDH]; // [pix][kloc]

    const int tid  = threadIdx.x;
    const int lane = tid & 31;
    const int wv   = tid >> 5;
    const int tile = blockIdx.x;          // 8 pixel tiles
    const int samp = blockIdx.y;          // 32 samples
    const int n0   = tile * 128;
    const int l16  = lane & 15;
    const int half = lane >> 4;

    v8f acc[8] = {};

    // fragment loads + 8 back-to-back WMMAs for one staged chunk
    auto compute = [&](int buf) {
        union F { struct { i4x lo, hi; } s; v16bf v; };
        F af;   // A frag: lane<16 K{0-7,16-23}, lane>=16 K{8-15,24-31}
        const unsigned short* ap = &As[buf][(wv * 16 + l16) * LDH + half * 8];
        af.s.lo = *(const i4x*)(ap);
        af.s.hi = *(const i4x*)(ap + 16);
        F bfr[8];
        #pragma unroll
        for (int nt = 0; nt < 8; ++nt) {
            const unsigned short* bp = &Bs[buf][(nt * 16 + l16) * LDH + half * 8];
            bfr[nt].s.lo = *(const i4x*)(bp);
            bfr[nt].s.hi = *(const i4x*)(bp + 16);
        }
        #pragma unroll
        for (int nt = 0; nt < 8; ++nt)
            acc[nt] = __builtin_amdgcn_wmma_f32_16x16x32_bf16(
                false, af.v, false, bfr[nt].v, (short)0, acc[nt], false, false);
    };

    if constexpr (PRE) {
        // per-thread tile-chunk coordinates (512 x 16B chunks for each of A and B)
        const int am = tid >> 2, aq = tid & 3;        // A: row m, quarter q (+64 rows for r=1)
#if HAVE_ASYNC
        auto stage_async = [&](int chunk, int buf) {
            const int tap = chunk >> 2;
            const int cb  = (chunk & 3) * 32;
            const int kh0 = tap / 3, kw0 = tap % 3;   // 0..2
            unsigned short* as = As[buf];
            unsigned short* bs = Bs[buf];
            const unsigned short* wsrc = wbf + (size_t)samp * 147456 + tap * 128 + cb;
            const unsigned short* xsrc = xT  + (size_t)samp * 147968 + cb;
            #pragma unroll
            for (int r = 0; r < 2; ++r) {
                const int m = am + r * 64;
                async_b128(&wsrc[(size_t)m * 1152 + aq * 8], &as[m * LDH + aq * 8]);
            }
            #pragma unroll
            for (int r = 0; r < 2; ++r) {
                const int n = am + r * 64;
                const int p = n0 + n;
                const size_t pos = (size_t)((p >> 5) + kh0) * 34 + ((p & 31) + kw0);
                async_b128(&xsrc[pos * 128 + aq * 8], &bs[n * LDH + aq * 8]);
            }
        };
        stage_async(0, 0);
        wait_async0();
        for (int chunk = 0; chunk < 36; ++chunk) {
            const int buf = chunk & 1;
            __syncthreads();
            if (chunk + 1 < 36) stage_async(chunk + 1, buf ^ 1);  // copy engine runs
            compute(buf);                                          // ... under the WMMAs
            if (chunk + 1 < 36) wait_async0();
        }
#else
        // register-staged fallback: global loads issue BEFORE compute, LDS stores AFTER,
        // so the WMMAs' dscnt waits never chain behind next-chunk global loads.
        struct SR { i4x a0, a1, b0, b1; };
        auto stage_load = [&](int chunk, SR& r) {
            const int tap = chunk >> 2;
            const int cb  = (chunk & 3) * 32;
            const int kh0 = tap / 3, kw0 = tap % 3;
            const unsigned short* wsrc = wbf + (size_t)samp * 147456 + tap * 128 + cb;
            const unsigned short* xsrc = xT  + (size_t)samp * 147968 + cb;
            r.a0 = *(const i4x*)&wsrc[(size_t)am * 1152 + aq * 8];
            r.a1 = *(const i4x*)&wsrc[(size_t)(am + 64) * 1152 + aq * 8];
            const int p0 = n0 + am, p1 = n0 + am + 64;
            const size_t q0 = (size_t)((p0 >> 5) + kh0) * 34 + ((p0 & 31) + kw0);
            const size_t q1 = (size_t)((p1 >> 5) + kh0) * 34 + ((p1 & 31) + kw0);
            r.b0 = *(const i4x*)&xsrc[q0 * 128 + aq * 8];
            r.b1 = *(const i4x*)&xsrc[q1 * 128 + aq * 8];
        };
        auto stage_store = [&](int buf, const SR& r) {
            unsigned short* as = As[buf];
            unsigned short* bs = Bs[buf];
            *(i4x*)&as[am * LDH + aq * 8]        = r.a0;
            *(i4x*)&as[(am + 64) * LDH + aq * 8] = r.a1;
            *(i4x*)&bs[am * LDH + aq * 8]        = r.b0;
            *(i4x*)&bs[(am + 64) * LDH + aq * 8] = r.b1;
        };
        SR r0; stage_load(0, r0); stage_store(0, r0);
        for (int chunk = 0; chunk < 36; ++chunk) {
            const int buf = chunk & 1;
            __syncthreads();
            SR rn;
            if (chunk + 1 < 36) stage_load(chunk + 1, rn);
            compute(buf);
            if (chunk + 1 < 36) stage_store(buf ^ 1, rn);
        }
#endif
    } else {
        // scratch-too-small fallback: fused f32->bf16 staging
        auto stage = [&](int chunk, int buf) {
            const int tap = chunk >> 2;
            const int cb  = (chunk & 3) * 32;
            const int kh = tap / 3 - 1, kw = tap % 3 - 1;
            unsigned short* as = As[buf];
            unsigned short* bs = Bs[buf];
            const float* wsrc = w + (size_t)samp * 147456;
            for (int idx = tid; idx < 128 * 32; idx += 256) {
                const int m = idx >> 5, kloc = idx & 31;
                as[m * LDH + kloc] = f2bf(wsrc[m * 1152 + (cb + kloc) * 9 + tap]);
            }
            const float* xsrc = x + (size_t)samp * 131072;
            for (int idx = tid; idx < 32 * 128; idx += 256) {
                const int kloc = idx >> 7, n = idx & 127;
                const int p = n0 + n;
                const int py = (p >> 5) + kh, px = (p & 31) + kw;
                float v = 0.0f;
                if ((unsigned)py < 32u && (unsigned)px < 32u)
                    v = xsrc[(cb + kloc) * 1024 + py * 32 + px];
                bs[n * LDH + kloc] = f2bf(v);
            }
        };
        stage(0, 0);
        for (int chunk = 0; chunk < 36; ++chunk) {
            const int buf = chunk & 1;
            __syncthreads();
            if (chunk + 1 < 36) stage(chunk + 1, buf ^ 1);
            compute(buf);
        }
    }

    // epilogue: D layout -> VGPR j: lanes 0-15 M=j, lanes 16-31 M=j+8; N = lane&15
    const float* bsp = bias + samp * 128;
    float* os = out + (size_t)samp * 131072;
    #pragma unroll
    for (int nt = 0; nt < 8; ++nt) {
        #pragma unroll
        for (int j = 0; j < 8; ++j) {
            const int ocr = wv * 16 + j + (half ? 8 : 0);
            const int pix = n0 + nt * 16 + l16;
            os[ocr * 1024 + pix] = acc[nt][j] + bsp[ocr];
        }
    }
}

extern "C" void kernel_launch(void* const* d_in, const int* in_sizes, int n_in,
                              void* d_out, int out_size, void* d_ws, size_t ws_size,
                              hipStream_t stream) {
    const float* x    = (const float*)d_in[0];   // [32][1][128][32][32]
    const float* wgt  = (const float*)d_in[1];   // [32][128][128][3][3]
    const float* bias = (const float*)d_in[2];   // [32][1][128]
    float* out        = (float*)d_out;

    const size_t wbf_elems = 32ull * 128 * 1152;   // 4,718,592  (bf16)
    const size_t xT_elems  = 32ull * 1156 * 128;   // 4,734,976  (bf16, padded NHWC)
    const size_t need = (wbf_elems + xT_elems) * sizeof(unsigned short); // ~18.9 MB

    dim3 grid(8, 32);
    dim3 block(256);

    if (ws_size >= need) {
        unsigned short* wbf = (unsigned short*)d_ws;
        unsigned short* xT  = wbf + wbf_elems;
        (void)hipMemsetAsync(xT, 0, xT_elems * sizeof(unsigned short), stream); // zero border
        prep_weight<<<dim3(4096), block, 0, stream>>>(wgt, wbf);
        prep_xT<<<dim3(4, 32), block, 0, stream>>>(x, xT);
        conv3x3_wmma<true><<<grid, block, 0, stream>>>(x, wgt, xT, wbf, bias, out);
    } else {
        conv3x3_wmma<false><<<grid, block, 0, stream>>>(x, wgt, nullptr, nullptr, bias, out);
    }
}
